// TprRnn_65335042507540
// MI455X (gfx1250) — matrix operand order repaired and checked by reading, not compiled, not run on hardware
//
#include <hip/hip_runtime.h>
#include <hip/hip_bf16.h>

typedef __attribute__((ext_vector_type(16))) _Float16 v16h;
typedef __attribute__((ext_vector_type(8)))  _Float16 v8h;
typedef __attribute__((ext_vector_type(8)))  float    v8f;

#define B_   128
#define S_   48
#define W_   12
#define V_   32000
#define SYM_ 128
#define H_   256
#define E_   48
#define R_   24
#define RP_  32    // R padded to 16-multiple for WMMA tiles
#define KXP_ 64    // x / Z contraction dim (48) padded to 32-multiple

// ---------------------------------------------------------------------------
// Weight prep: f32 [K,N] -> f16 transposed+padded [Np,Kp]  (Bt[n*Kp+k]=W[k,n])
// Zero-fills the padding so the WMMA fast path needs no bounds checks.
// ---------------------------------------------------------------------------
__global__ void cvt_bt_kernel(const float* __restrict__ in,
                              _Float16* __restrict__ out,
                              int K, int N, int Kp, int Np) {
  int i = blockIdx.x * blockDim.x + threadIdx.x;
  if (i >= Np * Kp) return;
  int n = i / Kp, k = i % Kp;
  float v = (n < N && k < K) ? in[k * N + n] : 0.f;
  out[i] = (_Float16)v;
}

// ---------------------------------------------------------------------------
// ssum[row,e] = sum_w word_embed[idx[row,w], e] * pos_embed[w, e]   (f16 out)
// ---------------------------------------------------------------------------
__global__ void embed_sum_kernel(const int* __restrict__ idx,
                                 const float* __restrict__ word_embed,
                                 const float* __restrict__ pos_embed,
                                 _Float16* __restrict__ out,
                                 int rows, int nwords) {
  int row = blockIdx.x;
  int e   = threadIdx.x;                 // blockDim.x == SYM_
  if (row >= rows) return;
  float s = 0.f;
  for (int w = 0; w < nwords; ++w) {
    int tok = idx[row * nwords + w];
    s += word_embed[tok * SYM_ + e] * pos_embed[w * SYM_ + e];
  }
  out[row * SYM_ + e] = (_Float16)s;
}

// ---------------------------------------------------------------------------
// WMMA GEMM: C[M,N] = act(A[M,Kp]_f16 @ Bt[Np,Kp]_f16^T + bias[N])
//   A  row-major, Kp % 32 == 0  (padded with zeros by producer)
//   Bt pre-transposed [n][k] with Np % 16 == 0, Kp % 32 == 0 (zero padded)
// One wave32 per 16x16 tile; 4 waves (4 adjacent N tiles) per block.
// Fragment fills are pure contiguous vector loads (2x b128 for A, 2x for B).
// Per CDNA5 ISA 7.12.2 (wave32):
//   A 16x32 f16 : lane m=L%16, half=L/16; elems [0..7]=K half*8..+7,
//                 [8..15]=K 16+half*8..+7  (two contiguous 8-half runs)
//   B 32x16 f16 : lane n=L%16, half=L/16; this code loads A and B alike: elem j -> K = half*16 + j
//   C 16x16 f32 : lane n=L%16, half=L/16; vgpr i -> M = half*8 + i
// ---------------------------------------------------------------------------
__global__ void wmma_gemm_kernel(const _Float16* __restrict__ A,
                                 const _Float16* __restrict__ Bt,
                                 const float* __restrict__ bias,
                                 float* __restrict__ Cf,      // f32 out (or null)
                                 _Float16* __restrict__ Ch,   // f16 out (or null)
                                 int M, int N, int Npad, int Kp, int act) {
  int ntn  = Npad >> 4;
  int ngrp = (ntn + 3) >> 2;
  int tm   = blockIdx.x / ngrp;
  int tg   = blockIdx.x % ngrp;
  int wv   = threadIdx.x >> 5;           // wave id within block (0..3)
  int tn   = tg * 4 + wv;
  if (tn >= ntn) return;                 // whole-wave early out, no barriers

  int lane = threadIdx.x & 31;
  int lm   = lane & 15;
  int half = lane >> 4;
  int m0   = tm << 4, n0 = tn << 4;

  const _Float16* arow = A  + (size_t)(m0 + lm) * Kp;
  const _Float16* bcol = Bt + (size_t)(n0 + lm) * Kp;

  v8f acc = {};
  for (int k0 = 0; k0 < Kp; k0 += 32) {
    __builtin_prefetch(arow + k0 + 64, 0, 1);   // global_prefetch_b8
    v16h a  = *(const v16h*)(arow + k0 + half * 16);
    v16h b  = *(const v16h*)(bcol + k0 + half * 16);
    acc = __builtin_amdgcn_wmma_f32_16x16x32_f16(
        /*neg_a=*/false, a, /*neg_b=*/false, b,
        /*c_mod=*/(short)0, acc, /*reuse_a=*/false, /*reuse_b=*/false);
  }

  int col = n0 + lm;
  if (col < N) {
    float bv = bias ? bias[col] : 0.f;
#pragma unroll
    for (int i = 0; i < 8; ++i) {
      int row = m0 + half * 8 + i;
      float v = acc[i] + bv;
      if (act) v = tanhf(v);
      if (Ch) Ch[row * N + col] = (_Float16)v;
      if (Cf) Cf[row * N + col] = v;
    }
  }
}

// ---------------------------------------------------------------------------
// Scan over S steps with the full TPR [E,R,F] resident in 221KB of LDS
// (CDNA5: 320KB/WGP), fused with the inference contractions + LayerNorms.
// One workgroup (256 threads = 8 waves) per batch element.
// ---------------------------------------------------------------------------
extern __shared__ float smem[];

__global__ void scan_infer_kernel(const float* __restrict__ e1g,
                                  const float* __restrict__ e2g,
                                  const float* __restrict__ r1g,
                                  const float* __restrict__ r2g,
                                  const float* __restrict__ r3g,
                                  const float* __restrict__ qe1g,
                                  const float* __restrict__ qr1g,
                                  const float* __restrict__ qr2g,
                                  const float* __restrict__ qr3g,
                                  const float* __restrict__ ln_gain,
                                  const float* __restrict__ ln_bias,
                                  _Float16* __restrict__ xout) {
  const int TPRN = E_ * R_ * E_;      // 55296
  float* TPR = smem;
  float* ve1 = TPR + TPRN;            // 48
  float* ve2 = ve1 + E_;              // 48
  float* vr1 = ve2 + E_;              // 24
  float* vr2 = vr1 + R_;              // 24
  float* vr3 = vr2 + R_;              // 24
  float* hat = vr3 + R_;              // 3*48 = 144 (w_hat | m_hat | b_hat)
  float* ent = hat + 144;             // 48: current entity vector
  float* xs  = ent + E_;              // 48: i1+i2+i3 accumulator

  int b  = blockIdx.x;
  int t  = threadIdx.x;
  int nt = blockDim.x;

  for (int i = t; i < TPRN; i += nt) TPR[i] = 0.f;
  __syncthreads();

  for (int s = 0; s < S_; ++s) {
    int rowE = (b * S_ + s) * E_;
    int rowR = (b * S_ + s) * R_;
    if (t < E_) { ve1[t] = e1g[rowE + t]; ve2[t] = e2g[rowE + t]; }
    else if (t < E_ + R_) {
      int r = t - E_;
      vr1[r] = r1g[rowR + r]; vr2[r] = r2g[rowR + r]; vr3[r] = r3g[rowR + r];
    }
    __syncthreads();

    // w_hat / m_hat / b_hat : 144 independent columns
    if (t < 144) {
      int h = t / E_, f = t % E_;
      const float* ev = (h == 2) ? ve2 : ve1;
      const float* rv = (h == 0) ? vr1 : (h == 1) ? vr2 : vr3;
      float acc = 0.f;
      for (int e = 0; e < E_; ++e) {
        const float* base = TPR + (e * R_) * E_ + f;
        float p = 0.f;
        for (int r = 0; r < R_; ++r) p += rv[r] * base[r * E_];
        acc += ev[e] * p;
      }
      hat[t] = acc;
    }
    __syncthreads();

    // TPR += e1 (x) [r1 (x) (e2 - w_hat) + r2 (x) (w_hat - m_hat)]
    //       + e2 (x) [r3 (x) (e1 - b_hat)]
    const float* w_h = hat;
    const float* m_h = hat + E_;
    const float* b_h = hat + 2 * E_;
    for (int i = t; i < TPRN; i += nt) {
      int e   = i / (R_ * E_);
      int rem = i % (R_ * E_);
      int r   = rem / E_;
      int f   = rem % E_;
      float upd = ve1[e] * (vr1[r] * (ve2[f] - w_h[f]) +
                            vr2[r] * (w_h[f] - m_h[f])) +
                  ve2[e] * (vr3[r] * (ve1[f] - b_h[f]));
      TPR[i] += upd;
    }
    __syncthreads();
  }

  // -------- inference: 3 chained contractions + LayerNorm --------
  if (t < E_) { ent[t] = qe1g[b * E_ + t]; xs[t] = 0.f; }
  __syncthreads();
  const float* qrs[3] = { qr1g, qr2g, qr3g };
  for (int step = 0; step < 3; ++step) {
    if (t < R_) vr1[t] = qrs[step][b * R_ + t];
    __syncthreads();
    if (t < E_) {
      float acc = 0.f;
      for (int e = 0; e < E_; ++e) {
        const float* base = TPR + (e * R_) * E_ + t;
        float p = 0.f;
        for (int r = 0; r < R_; ++r) p += vr1[r] * base[r * E_];
        acc += ent[e] * p;
      }
      hat[t] = acc;
    }
    __syncthreads();
    if (t == 0) {
      float mu = 0.f;
      for (int f = 0; f < E_; ++f) mu += hat[f];
      mu /= (float)E_;
      float var = 0.f;
      for (int f = 0; f < E_; ++f) { float d = hat[f] - mu; var += d * d; }
      var /= (float)E_;
      hat[E_] = mu;
      hat[E_ + 1] = rsqrtf(var + 1e-6f);
    }
    __syncthreads();
    if (t < E_) {
      float v = ln_gain[step * E_ + t] * (hat[t] - hat[E_]) * hat[E_ + 1] +
                ln_bias[step * E_ + t];
      ent[t] = v;                  // entity vector for next contraction
      xs[t] += v;
    }
    __syncthreads();
  }
  // x padded to KXP_ (=64) columns of f16 for the final WMMA GEMM
  if (t < KXP_) xout[b * KXP_ + t] = (_Float16)((t < E_) ? xs[t] : 0.f);
}

// ---------------------------------------------------------------------------
extern "C" void kernel_launch(void* const* d_in, const int* in_sizes, int n_in,
                              void* d_out, int out_size, void* d_ws, size_t ws_size,
                              hipStream_t stream) {
  (void)in_sizes; (void)n_in; (void)out_size; (void)ws_size;

  const int*   story      = (const int*)d_in[0];
  const int*   query      = (const int*)d_in[1];
  const float* word_embed = (const float*)d_in[2];
  const float* pos_embed  = (const float*)d_in[3];
  const float* Z          = (const float*)d_in[4];
  const float* ue_W1 = (const float*)d_in[5];
  const float* ue_b1 = (const float*)d_in[6];
  const float* ue_W2 = (const float*)d_in[7];
  const float* ue_b2 = (const float*)d_in[8];
  const float* ur_W1 = (const float*)d_in[9];
  const float* ur_b1 = (const float*)d_in[10];
  const float* ur_W2 = (const float*)d_in[11];
  const float* ur_b2 = (const float*)d_in[12];
  const float* ie_W1 = (const float*)d_in[13];
  const float* ie_b1 = (const float*)d_in[14];
  const float* ie_W2 = (const float*)d_in[15];
  const float* ie_b2 = (const float*)d_in[16];
  const float* ir_W1 = (const float*)d_in[17];
  const float* ir_b1 = (const float*)d_in[18];
  const float* ir_W2 = (const float*)d_in[19];
  const float* ir_b2 = (const float*)d_in[20];
  const float* ln_g  = (const float*)d_in[21];
  const float* ln_b  = (const float*)d_in[22];
  float* logits = (float*)d_out;

  // ---- workspace carve ----
  size_t off = 0;
  auto carve = [&](size_t bytes) -> void* {
    void* p = (char*)d_ws + off;
    off += (bytes + 255) & ~(size_t)255;
    return p;
  };
  const int ROWS = B_ * S_;
  _Float16* ssum    = (_Float16*)carve((size_t)ROWS * SYM_ * 2);
  _Float16* qsum    = (_Float16*)carve((size_t)B_ * SYM_ * 2);
  _Float16* hidden  = (_Float16*)carve((size_t)ROWS * H_ * 2);
  _Float16* qhidden = (_Float16*)carve((size_t)B_ * H_ * 2);
  float* e1 = (float*)carve((size_t)ROWS * E_ * 4);
  float* e2 = (float*)carve((size_t)ROWS * E_ * 4);
  float* r1 = (float*)carve((size_t)ROWS * R_ * 4);
  float* r2 = (float*)carve((size_t)ROWS * R_ * 4);
  float* r3 = (float*)carve((size_t)ROWS * R_ * 4);
  float* qe1 = (float*)carve((size_t)B_ * E_ * 4);
  float* qr1 = (float*)carve((size_t)B_ * R_ * 4);
  float* qr2 = (float*)carve((size_t)B_ * R_ * 4);
  float* qr3 = (float*)carve((size_t)B_ * R_ * 4);
  // pre-transposed + padded f16 weights: Bt[Np][Kp]
  _Float16* ue_W1bt = (_Float16*)carve((size_t)2 * H_ * SYM_ * 2);   // [256][128] x2
  _Float16* ur_W1bt = (_Float16*)carve((size_t)3 * H_ * SYM_ * 2);   // [256][128] x3
  _Float16* ue_W2bt = (_Float16*)carve((size_t)2 * E_ * H_ * 2);     // [48][256]  x2
  _Float16* ur_W2bt = (_Float16*)carve((size_t)3 * RP_ * H_ * 2);    // [32][256]  x3
  _Float16* ie_W1bt = (_Float16*)carve((size_t)H_ * SYM_ * 2);       // head 0 only
  _Float16* ir_W1bt = (_Float16*)carve((size_t)3 * H_ * SYM_ * 2);
  _Float16* ie_W2bt = (_Float16*)carve((size_t)E_ * H_ * 2);
  _Float16* ir_W2bt = (_Float16*)carve((size_t)3 * RP_ * H_ * 2);
  _Float16* Zbt     = (_Float16*)carve((size_t)V_ * KXP_ * 2);       // [32000][64]
  _Float16* xh      = (_Float16*)carve((size_t)B_ * KXP_ * 2);       // [128][64]

  auto cvtbt = [&](const float* src, _Float16* dst, int K, int N, int Kp, int Np) {
    int n = Np * Kp;
    cvt_bt_kernel<<<(n + 255) / 256, 256, 0, stream>>>(src, dst, K, N, Kp, Np);
  };
  for (int h = 0; h < 2; ++h)
    cvtbt(ue_W1 + h * SYM_ * H_, ue_W1bt + h * H_ * SYM_, SYM_, H_, SYM_, H_);
  for (int h = 0; h < 3; ++h)
    cvtbt(ur_W1 + h * SYM_ * H_, ur_W1bt + h * H_ * SYM_, SYM_, H_, SYM_, H_);
  for (int h = 0; h < 2; ++h)
    cvtbt(ue_W2 + h * H_ * E_, ue_W2bt + h * E_ * H_, H_, E_, H_, E_);
  for (int h = 0; h < 3; ++h)
    cvtbt(ur_W2 + h * H_ * R_, ur_W2bt + h * RP_ * H_, H_, R_, H_, RP_);
  cvtbt(ie_W1, ie_W1bt, SYM_, H_, SYM_, H_);
  for (int h = 0; h < 3; ++h)
    cvtbt(ir_W1 + h * SYM_ * H_, ir_W1bt + h * H_ * SYM_, SYM_, H_, SYM_, H_);
  cvtbt(ie_W2, ie_W2bt, H_, E_, H_, E_);
  for (int h = 0; h < 3; ++h)
    cvtbt(ir_W2 + h * H_ * R_, ir_W2bt + h * RP_ * H_, H_, R_, H_, RP_);
  cvtbt(Z, Zbt, E_, V_, KXP_, V_);

  embed_sum_kernel<<<ROWS, SYM_, 0, stream>>>(story, word_embed, pos_embed,
                                              ssum, ROWS, W_);
  embed_sum_kernel<<<B_, SYM_, 0, stream>>>(query, word_embed, pos_embed,
                                            qsum, B_, W_);

  auto gemm = [&](const _Float16* A, const _Float16* Bt, const float* bias,
                  float* Cf, _Float16* Ch, int M, int N, int Npad, int Kp,
                  int act) {
    int ntn  = Npad >> 4;
    int ngrp = (ntn + 3) >> 2;
    int blocks = (M / 16) * ngrp;
    wmma_gemm_kernel<<<blocks, 128, 0, stream>>>(A, Bt, bias, Cf, Ch,
                                                 M, N, Npad, Kp, act);
  };

  // Update-module MLP heads (M = 6144)
  float* eouts[2] = { e1, e2 };
  for (int h = 0; h < 2; ++h) {
    gemm(ssum, ue_W1bt + h * H_ * SYM_, ue_b1 + h * H_, nullptr, hidden,
         ROWS, H_, H_, SYM_, 1);
    gemm(hidden, ue_W2bt + h * E_ * H_, ue_b2 + h * E_, eouts[h], nullptr,
         ROWS, E_, E_, H_, 1);
  }
  float* routs[3] = { r1, r2, r3 };
  for (int h = 0; h < 3; ++h) {
    gemm(ssum, ur_W1bt + h * H_ * SYM_, ur_b1 + h * H_, nullptr, hidden,
         ROWS, H_, H_, SYM_, 1);
    gemm(hidden, ur_W2bt + h * RP_ * H_, ur_b2 + h * R_, routs[h], nullptr,
         ROWS, R_, RP_, H_, 1);
  }

  // Inference-module MLP heads (M = 128); only ie head 0 is used.
  gemm(qsum, ie_W1bt, ie_b1, nullptr, qhidden, B_, H_, H_, SYM_, 1);
  gemm(qhidden, ie_W2bt, ie_b2, qe1, nullptr, B_, E_, E_, H_, 1);
  float* qrouts[3] = { qr1, qr2, qr3 };
  for (int h = 0; h < 3; ++h) {
    gemm(qsum, ir_W1bt + h * H_ * SYM_, ir_b1 + h * H_, nullptr, qhidden,
         B_, H_, H_, SYM_, 1);
    gemm(qhidden, ir_W2bt + h * RP_ * H_, ir_b2 + h * R_, qrouts[h], nullptr,
         B_, R_, RP_, H_, 1);
  }

  // Scan + inference with LDS-resident TPR (221KB < 320KB/WGP)
  const int SMEM_FLOATS = E_ * R_ * E_ + 2 * E_ + 3 * R_ + 144 + 2 * E_;
  const int SMEM_BYTES  = SMEM_FLOATS * 4;
  hipFuncSetAttribute((const void*)scan_infer_kernel,
                      hipFuncAttributeMaxDynamicSharedMemorySize, SMEM_BYTES);
  scan_infer_kernel<<<B_, 256, SMEM_BYTES, stream>>>(
      e1, e2, r1, r2, r3, qe1, qr1, qr2, qr3, ln_g, ln_b, xh);

  // logits[128, 32000] = x @ Z
  gemm(xh, Zbt, nullptr, logits, nullptr, B_, V_, V_, KXP_, 0);
}
